// LinearAttention_4587025072774
// MI455X (gfx1250) — matrix operand order repaired and checked by reading, not compiled, not run on hardware
//
#include <hip/hip_runtime.h>

// ---------------------------------------------------------------------------
// LinearAttention (relu attention, no softmax) for MI455X / gfx1250 (wave32)
//   Q = q @ Wq^T + bq ; K = k @ Wk^T + bk ; V = v @ Wv^T + bv
//   out = relu(Q K^T) @ V
// GEMMs on v_wmma_f32_16x16x32_bf16; bf16-operand GEMMs stage tiles with the
// Tensor Data Mover (tensor_load_to_lds + s_wait_tensorcnt), double-buffered.
// ---------------------------------------------------------------------------

typedef __bf16 bf16_t;
typedef bf16_t v16bf __attribute__((ext_vector_type(16)));
typedef float  v8f   __attribute__((ext_vector_type(8)));
typedef unsigned int v4u __attribute__((ext_vector_type(4)));
typedef int          v8i __attribute__((ext_vector_type(8)));
typedef int          v4i __attribute__((ext_vector_type(4)));

#define BDIM 4
#define SDIM 2048
#define DDIM 1024

#define LDS_STRIDE 40   // 32 bf16 data + 8 pad = 80B rows: 16B aligned, bank-conflict free

union FragU { v16bf v; uint4 u[2]; };
union Pack8 { bf16_t h[8]; uint4 u; };

// A fragment (16x32 bf16, M x K), ISA 7.12.2: lane row = l&15;
// lanes 0-15 hold K {0..7,16..23}, lanes 16-31 hold K {8..15,24..31}.
__device__ __forceinline__ v16bf ldsA_frag(const bf16_t* tile, int lrow, int lhalf) {
  const bf16_t* row = tile + lrow * LDS_STRIDE;
  const int kb = lhalf * 8;
  FragU f;
  f.u[0] = *(const uint4*)(row + kb);
  f.u[1] = *(const uint4*)(row + 16 + kb);
  return f.v;
}

// B fragment (32x16 bf16, K x N) read from a Bt (N-major) LDS tile:
// lane col n = l&15; lanes 0-15 hold K {0..7,16..23}, lanes 16-31 hold K {8..15,24..31}.
__device__ __forceinline__ v16bf ldsB_frag(const bf16_t* tile, int lrow, int lhalf) {
  const bf16_t* row = tile + lrow * LDS_STRIDE;
  const int kb = lhalf * 8;
  FragU f;
  f.u[0] = *(const uint4*)(row + kb);
  f.u[1] = *(const uint4*)(row + 16 + kb);
  return f.v;
}

__device__ __forceinline__ v8f wmma_bf16(v16bf a, v16bf b, v8f c) {
  return __builtin_amdgcn_wmma_f32_16x16x32_bf16(
      /*neg_a=*/false, a, /*neg_b=*/false, b,
      /*c_mod=*/(short)0, c, /*reuse_a=*/false, /*reuse_b=*/false);
}

// ---------------------------------------------------------------------------
// Tensor Data Mover: DMA one [rows x 32] bf16 tile (row stride = ldElems bf16)
// from global to LDS at lds_off, inserting 4 dwords of pad every 16 dwords so
// the LDS image has the same 80B padded row stride as the manual path.
// D# packing per CDNA5 ISA sec 8.3/8.4 (group2/3 zero => 2D tile).
// ---------------------------------------------------------------------------
__device__ __forceinline__ void tdm_load_tile(const bf16_t* gsrc, unsigned lds_off,
                                              unsigned rows, unsigned ldElems,
                                              unsigned tensorRows) {
  const unsigned long long ga = (unsigned long long)(uintptr_t)gsrc;
  v4u g0;
  g0[0] = 1u;                                        // count=1, user descriptor
  g0[1] = lds_off;                                   // lds_addr
  g0[2] = (unsigned)(ga & 0xffffffffu);              // global_addr[31:0]
  g0[3] = (unsigned)((ga >> 32) & 0x01ffffffu) | (2u << 30);  // addr[56:32] | type=2

  const unsigned dim0 = ldElems;                     // tensor_dim0 (elems, OOB bound)
  const unsigned dim1 = tensorRows;                  // tensor_dim1 (rows, OOB bound)
  v8i g1;
  g1[0] = (int)((1u << 16)        // data_size = 2 bytes
              | (1u << 20)        // pad_enable
              | (3u << 22)        // pad_interval: 16 dwords (64B) between pads
              | (3u << 25));      // pad_amount:   4 dwords (16B)
  g1[1] = (int)((dim0 & 0xffffu) << 16);                       // tensor_dim0[15:0]
  g1[2] = (int)((dim0 >> 16) | ((dim1 & 0xffffu) << 16));      // dim0[31:16]|dim1[15:0]
  g1[3] = (int)((dim1 >> 16) | (32u << 16));                   // dim1[31:16]|tile_dim0=32
  g1[4] = (int)rows;                                           // tile_dim1=rows, tile_dim2=0
  g1[5] = (int)dim0;                                           // tensor_dim0_stride[31:0]
  g1[6] = 0;                                                   // stride[47:32] | dim1_stride lo
  g1[7] = 0;
  const v4i z4 = {0, 0, 0, 0};
  const v8i z8 = {0, 0, 0, 0, 0, 0, 0, 0};
  __builtin_amdgcn_tensor_load_to_lds(g0, g1, z4, z4, z8, 0);
}

// C = A[M,K] (row-major) x Bt[N,K]^T  (Bt row-major = B^T), optional bias/relu,
// optional transposed bf16 store (for V^T), f32 or bf16 output.
// lda == ldb == K for every GEMM in this pipeline.
template<bool A_F32, bool B_F32, bool BIAS, bool RELU, bool TRANS, bool OUT_F32, bool USE_TDM>
__global__ __launch_bounds__(256)
void gemm_wmma(const void* __restrict__ Aall, const void* __restrict__ Ball,
               const float* __restrict__ bias, void* __restrict__ Out,
               int M, int N, int K,
               long aBatch, long bBatch, long oBatch, int trans_s)
{
  __shared__ __align__(16) bf16_t smA[2][64  * LDS_STRIDE];
  __shared__ __align__(16) bf16_t smB[2][128 * LDS_STRIDE];

  const int tid   = threadIdx.x;
  const int lane  = tid & 31;
  const int wave  = tid >> 5;
  const int lrow  = lane & 15;
  const int lhalf = lane >> 4;
  const int wm    = wave & 1;   // 2 waves along M  (32 rows each)
  const int wn    = wave >> 1;  // 4 waves along N  (32 cols each)

  const int bm = blockIdx.x * 64;
  const int bn = blockIdx.y * 128;
  const int bz = blockIdx.z;

  const float*  Af = (const float*) Aall + (size_t)bz * aBatch;
  const bf16_t* Ah = (const bf16_t*)Aall + (size_t)bz * aBatch;
  const float*  Bf = (const float*) Ball + (size_t)bz * bBatch;
  const bf16_t* Bh = (const bf16_t*)Ball + (size_t)bz * bBatch;

  v8f acc[2][2] = {};

  if constexpr (USE_TDM) {
    // ---- bf16 operands: tiles staged by the Tensor Data Mover, 2x buffered ---
    const bf16_t* Atile = Ah + (size_t)bm * K;   // [64 x K] slab, tile col = k0
    const bf16_t* Btile = Bh + (size_t)bn * K;   // [128 x K] slab
    const int steps = K / 32;

    if (wave == 0) {
      tdm_load_tile(Atile, (unsigned)(uintptr_t)&smA[0][0], 64u,  (unsigned)K, 64u);
      tdm_load_tile(Btile, (unsigned)(uintptr_t)&smB[0][0], 128u, (unsigned)K, 128u);
    }
    for (int i = 0; i < steps; ++i) {
      const int p = i & 1;
      if (wave == 0) __builtin_amdgcn_s_wait_tensorcnt(0);   // tiles i landed
      __syncthreads();                                       // publish LDS; also fences
                                                             // last reads of buffer p^1
      if (wave == 0 && (i + 1) < steps) {                    // DMA tiles i+1 while computing
        const size_t koff = (size_t)(i + 1) * 32;
        tdm_load_tile(Atile + koff, (unsigned)(uintptr_t)&smA[p ^ 1][0], 64u,  (unsigned)K, 64u);
        tdm_load_tile(Btile + koff, (unsigned)(uintptr_t)&smB[p ^ 1][0], 128u, (unsigned)K, 128u);
      }
      const bf16_t* tileA = &smA[p][0] + (wm * 32) * LDS_STRIDE;
      const bf16_t* tileB = &smB[p][0] + (wn * 32) * LDS_STRIDE;
      v16bf a0 = ldsA_frag(tileA,                   lrow, lhalf);
      v16bf a1 = ldsA_frag(tileA + 16 * LDS_STRIDE, lrow, lhalf);
      v16bf b0 = ldsB_frag(tileB,                   lrow, lhalf);
      v16bf b1 = ldsB_frag(tileB + 16 * LDS_STRIDE, lrow, lhalf);
      acc[0][0] = wmma_bf16(a0, b0, acc[0][0]);
      acc[0][1] = wmma_bf16(a0, b1, acc[0][1]);
      acc[1][0] = wmma_bf16(a1, b0, acc[1][0]);
      acc[1][1] = wmma_bf16(a1, b1, acc[1][1]);
    }
  } else {
    // ---- f32 operands: manual stage + convert (v_cvt_pk_bf16_f32) ----------
    const int lr = tid >> 2;        // 0..63
    const int lc = (tid & 3) * 8;   // 0,8,16,24

    for (int k0 = 0; k0 < K; k0 += 32) {
      {
        const float* g = Af + (size_t)(bm + lr) * K + k0 + lc;
        float4 x0 = ((const float4*)g)[0];
        float4 x1 = ((const float4*)g)[1];
        Pack8 pk;
        pk.h[0] = (bf16_t)x0.x; pk.h[1] = (bf16_t)x0.y;
        pk.h[2] = (bf16_t)x0.z; pk.h[3] = (bf16_t)x0.w;
        pk.h[4] = (bf16_t)x1.x; pk.h[5] = (bf16_t)x1.y;
        pk.h[6] = (bf16_t)x1.z; pk.h[7] = (bf16_t)x1.w;
        *(uint4*)&smA[0][lr * LDS_STRIDE + lc] = pk.u;
        if (k0 + 32 < K) __builtin_prefetch(g + 32, 0, 3);   // global_prefetch_b8
      }
#pragma unroll
      for (int pp = 0; pp < 2; ++pp) {
        const int rr = lr + pp * 64;
        const float* g = Bf + (size_t)(bn + rr) * K + k0 + lc;
        float4 x0 = ((const float4*)g)[0];
        float4 x1 = ((const float4*)g)[1];
        Pack8 pk;
        pk.h[0] = (bf16_t)x0.x; pk.h[1] = (bf16_t)x0.y;
        pk.h[2] = (bf16_t)x0.z; pk.h[3] = (bf16_t)x0.w;
        pk.h[4] = (bf16_t)x1.x; pk.h[5] = (bf16_t)x1.y;
        pk.h[6] = (bf16_t)x1.z; pk.h[7] = (bf16_t)x1.w;
        *(uint4*)&smB[0][rr * LDS_STRIDE + lc] = pk.u;
        if (k0 + 32 < K) __builtin_prefetch(g + 32, 0, 3);
      }
      __syncthreads();

      const bf16_t* tileA = &smA[0][0] + (wm * 32) * LDS_STRIDE;
      const bf16_t* tileB = &smB[0][0] + (wn * 32) * LDS_STRIDE;
      v16bf a0 = ldsA_frag(tileA,                   lrow, lhalf);
      v16bf a1 = ldsA_frag(tileA + 16 * LDS_STRIDE, lrow, lhalf);
      v16bf b0 = ldsB_frag(tileB,                   lrow, lhalf);
      v16bf b1 = ldsB_frag(tileB + 16 * LDS_STRIDE, lrow, lhalf);
      acc[0][0] = wmma_bf16(a0, b0, acc[0][0]);
      acc[0][1] = wmma_bf16(a0, b1, acc[0][1]);
      acc[1][0] = wmma_bf16(a1, b0, acc[1][0]);
      acc[1][1] = wmma_bf16(a1, b1, acc[1][1]);
      __syncthreads();
    }
  }

  // ---- store: C/D layout lane l holds (M = g + 8*(l>>4), N = l&15) ----
  float*  Of = (float*) Out + (size_t)bz * oBatch;
  bf16_t* Oh = (bf16_t*)Out + (size_t)bz * oBatch;
#pragma unroll
  for (int tm = 0; tm < 2; ++tm) {
#pragma unroll
    for (int tn = 0; tn < 2; ++tn) {
#pragma unroll
      for (int g = 0; g < 8; ++g) {
        const int m = bm + wm * 32 + tm * 16 + g + lhalf * 8;
        const int n = bn + wn * 32 + tn * 16 + lrow;
        float v = acc[tm][tn][g];
        if (BIAS) v += bias[n];
        if (RELU) v = fmaxf(v, 0.0f);
        if (OUT_F32) {
          Of[(size_t)m * N + n] = v;
        } else if (TRANS) {
          // store V projection as Vt[b][n(d)][s]
          const int bb = m / trans_s;
          const int ss = m - bb * trans_s;
          Oh[((size_t)bb * N + n) * (size_t)trans_s + ss] = (bf16_t)v;
        } else {
          Oh[(size_t)m * N + n] = (bf16_t)v;
        }
      }
    }
  }
}

extern "C" void kernel_launch(void* const* d_in, const int* in_sizes, int n_in,
                              void* d_out, int out_size, void* d_ws, size_t ws_size,
                              hipStream_t stream) {
  (void)in_sizes; (void)n_in; (void)out_size; (void)ws_size;

  const float* queries = (const float*)d_in[0];
  const float* keys    = (const float*)d_in[1];
  const float* values  = (const float*)d_in[2];
  const float* Wq      = (const float*)d_in[3];
  const float* bq      = (const float*)d_in[4];
  const float* Wk      = (const float*)d_in[5];
  const float* bk      = (const float*)d_in[6];
  const float* Wv      = (const float*)d_in[7];
  const float* bv      = (const float*)d_in[8];

  const size_t projElems = (size_t)BDIM * SDIM * DDIM;  // 8.39M bf16
  bf16_t* Qp = (bf16_t*)d_ws;                           // [B*S, D]
  bf16_t* Kp = Qp + projElems;                          // [B*S, D]
  bf16_t* Vt = Kp + projElems;                          // [B, D, S]
  bf16_t* Sc = Vt + projElems;                          // [B, S, S] relu scores

  const dim3 blk(256);
  const int  Mflat = BDIM * SDIM;  // 8192

  // --- projections:  C[M=8192, N=1024] = X @ W^T + b,   bf16 out ---
  gemm_wmma<true, true, true, false, false, false, false>
      <<<dim3(Mflat / 64, DDIM / 128, 1), blk, 0, stream>>>(
          queries, Wq, bq, Qp, Mflat, DDIM, DDIM, 0, 0, 0, SDIM);
  gemm_wmma<true, true, true, false, false, false, false>
      <<<dim3(Mflat / 64, DDIM / 128, 1), blk, 0, stream>>>(
          keys, Wk, bk, Kp, Mflat, DDIM, DDIM, 0, 0, 0, SDIM);
  gemm_wmma<true, true, true, false, true, false, false>     // transposed store -> Vt
      <<<dim3(Mflat / 64, DDIM / 128, 1), blk, 0, stream>>>(
          values, Wv, bv, Vt, Mflat, DDIM, DDIM, 0, 0, 0, SDIM);

  // --- scores: per batch  P[S,S] = relu(Qp @ Kp^T),  bf16 out, TDM staging ---
  gemm_wmma<false, false, false, true, false, false, true>
      <<<dim3(SDIM / 64, SDIM / 128, BDIM), blk, 0, stream>>>(
          Qp, Kp, nullptr, Sc, SDIM, SDIM, DDIM,
          (long)SDIM * DDIM, (long)SDIM * DDIM, (long)SDIM * SDIM, SDIM);

  // --- output: per batch  O[S,D] = P @ V (Bt = Vt row-major), f32, TDM staging ---
  gemm_wmma<false, false, false, false, false, true, true>
      <<<dim3(SDIM / 64, DDIM / 128, BDIM), blk, 0, stream>>>(
          Sc, Vt, nullptr, d_out, SDIM, DDIM, SDIM,
          (long)SDIM * SDIM, (long)DDIM * SDIM, (long)SDIM * DDIM, SDIM);
}
